// RAMLayerOriginal_21818433864468
// MI455X (gfx1250) — 8-bit matrix operand order repaired and checked by reading, compiled, not run on hardware
//
#include <hip/hip_runtime.h>
#include <stdint.h>

#define T_DIM 4096
#define N_DIM 2048
#define K_DIM 16
#define WORDS_PER_ROW (T_DIM / 32)
#define BPB   4
#define NPB   256
#define WPB   8

typedef int v8i __attribute__((ext_vector_type(8)));

__global__ __launch_bounds__(256) void pack_bits_kernel(
    const int* __restrict__ in, uint32_t* __restrict__ packed)
{
    int gid = blockIdx.x * 256 + threadIdx.x;
    bool bit = (in[gid] != 0);
#if __has_builtin(__builtin_amdgcn_ballot_w32)
    uint32_t msk = __builtin_amdgcn_ballot_w32(bit);
#else
    uint32_t msk = (uint32_t)__ballot(bit);
#endif
    if ((threadIdx.x & 31) == 0)
        packed[gid >> 5] = msk;
}

__global__ __launch_bounds__(256) void ram_lookup_kernel(
    const uint32_t* __restrict__ packed,
    const int*      __restrict__ connections,
    const int*      __restrict__ memory,
    float*          __restrict__ out)
{
    __shared__ uint32_t s_bits[BPB * WORDS_PER_ROW];
    __shared__ uint32_t s_d[WPB][2][4][8];

    const int tid   = threadIdx.x;
    const int lane  = tid & 31;
    const int w     = tid >> 5;
    const int b0    = blockIdx.x * BPB;
    const int nbase = blockIdx.y * NPB + w * 32;

    if (tid < (BPB * WORDS_PER_ROW) / 4) {
        uint32_t lds_off = (uint32_t)(uintptr_t)(&s_bits[tid * 4]);
        uint64_t src = (uint64_t)(uintptr_t)(packed + (size_t)b0 * WORDS_PER_ROW + tid * 4);
        asm volatile("global_load_async_to_lds_b128 %0, %1, off"
                     :: "v"(lds_off), "v"(src) : "memory");
        asm volatile("s_wait_asynccnt 0" ::: "memory");
    }

    const int h  = lane >> 4;
    const int m  = lane & 15;
    const int n0 = nbase + m;
    const int n1 = nbase + 16 + m;
    const int4* cp0 = (const int4*)(connections + (size_t)n0 * K_DIM + h * 8);
    const int4* cp1 = (const int4*)(connections + (size_t)n1 * K_DIM + h * 8);
    const int4 c00 = cp0[0], c01 = cp0[1];
    const int4 c10 = cp1[0], c11 = cp1[1];

    v8i bm = {};
    if (lane == 0) { bm[0] = 0x08040201; bm[1] = (int)0x80402010u; }
    if (lane == 17) { bm[0] = 0x08040201; bm[1] = (int)0x80402010u; }

    __syncthreads();

    const int g_out = lane >> 4;
    const int n_out = nbase + g_out * 16 + m;
    const int* memrow = memory + ((size_t)n_out << 16);

    for (int bi = 0; bi < BPB; ++bi) {
        const uint32_t* bits = &s_bits[bi * WORDS_PER_ROW];

        #pragma unroll
        for (int g = 0; g < 2; ++g) {
            const int4 ca = (g == 0) ? c00 : c10;
            const int4 cb = (g == 0) ? c01 : c11;
            uint32_t a0 = 0, a1 = 0;
            int idx;
            idx = ca.x; a0 |=  (bits[idx >> 5] >> (idx & 31)) & 1u;
            idx = ca.y; a0 |= ((bits[idx >> 5] >> (idx & 31)) & 1u) << 8;
            idx = ca.z; a0 |= ((bits[idx >> 5] >> (idx & 31)) & 1u) << 16;
            idx = ca.w; a0 |= ((bits[idx >> 5] >> (idx & 31)) & 1u) << 24;
            idx = cb.x; a1 |=  (bits[idx >> 5] >> (idx & 31)) & 1u;
            idx = cb.y; a1 |= ((bits[idx >> 5] >> (idx & 31)) & 1u) << 8;
            idx = cb.z; a1 |= ((bits[idx >> 5] >> (idx & 31)) & 1u) << 16;
            idx = cb.w; a1 |= ((bits[idx >> 5] >> (idx & 31)) & 1u) << 24;

            v8i am = {};
            am[0] = (int)a0; am[1] = (int)a1;
            v8i cm = {};
            v8i d = __builtin_amdgcn_wmma_i32_16x16x64_iu8(
                        false, am, false, bm, cm, false, false);

            if (lane == 0 || lane == 1 || lane == 16 || lane == 17) {
                const int slot = ((lane & 1) << 1) | (lane >> 4);
                #pragma unroll
                for (int r = 0; r < 8; ++r)
                    s_d[w][g][slot][r] = (uint32_t)d[r];
            }
        }
        __syncthreads();

        {
            const uint32_t lo = s_d[w][g_out][(m < 8) ? 0 : 1][m & 7];
            const uint32_t hi = s_d[w][g_out][(m < 8) ? 2 : 3][m & 7];
            const uint32_t addr = lo | (hi << 8);
            const int bb = b0 + bi;
            const float v = (float)(memrow[addr] & 1);
            __builtin_nontemporal_store(v, &out[(size_t)bb * N_DIM + n_out]);
        }
        __syncthreads();
    }
}

extern "C" void kernel_launch(void* const* d_in, const int* in_sizes, int n_in,
                              void* d_out, int out_size, void* d_ws, size_t ws_size,
                              hipStream_t stream) {
    const int* input_bits  = (const int*)d_in[0];
    const int* connections = (const int*)d_in[1];
    const int* memory      = (const int*)d_in[2];
    float*     out         = (float*)d_out;
    uint32_t*  packed      = (uint32_t*)d_ws;

    const int B = in_sizes[0] / T_DIM;
    const int total = B * T_DIM;

    pack_bits_kernel<<<dim3(total / 256), 256, 0, stream>>>(input_bits, packed);

    dim3 grid(B / BPB, N_DIM / NPB);
    ram_lookup_kernel<<<grid, 256, 0, stream>>>(packed, connections, memory, out);
}
